// SimpleTransformer_91096256348597
// MI455X (gfx1250) — compile-verified, matrix operand order checked by reading, not run on hardware
//
#include <hip/hip_runtime.h>
#include <hip/hip_bf16.h>

typedef __bf16 bf16;
typedef __attribute__((ext_vector_type(16))) __bf16 v16bf;
typedef __attribute__((ext_vector_type(8)))  __bf16 v8bf;
typedef __attribute__((ext_vector_type(8)))  float  v8f;

#define SEQ   4097
#define SP    4224   // padded rows: multiple of 128
#define HDIM  1024
#define NHEADS 16

// ---------------------------------------------------------------------------
// WMMA operand fragment load (A or B operand, 16-bit data, 16x32 / [N=16]x32).
// ISA 7.12.2: lane m=lane&15 holds 16 K-values: elems 0..7 -> K = g+0..7,
// elems 8..15 -> K = 16+g+0..7 where g = (lane>>4)*8. Two aligned 16B loads.
// ---------------------------------------------------------------------------
__device__ __forceinline__ v16bf load_frag(const bf16* base, int ld, int row0,
                                           int col0, int lane) {
  const int m = lane & 15;
  const int g = (lane >> 4) << 3;
  const bf16* p = base + (row0 + m) * ld + col0 + g;
  v8bf lo = *(const v8bf*)(p);
  v8bf hi = *(const v8bf*)(p + 16);
  return __builtin_shufflevector(lo, hi, 0,1,2,3,4,5,6,7,8,9,10,11,12,13,14,15);
}

// ---------------------------------------------------------------------------
// Generic GEMM: C[M,N] = act(A[M,K] * W[N,K]^T + bias [+ resid])
// 128x128 block tile, 256 threads = 8 waves, each wave 64x32 (4x2 fragments).
// Software-pipelined: double-buffered LDS, next K-tile loaded into registers
// during the current tile's WMMAs; one barrier per K-step.
// ---------------------------------------------------------------------------
__global__ __launch_bounds__(256) void gemm_kernel(
    const float* __restrict__ A, const float* __restrict__ W,
    const float* __restrict__ bias, const float* __restrict__ resid,
    float* __restrict__ C, int M, int N, int K, int relu)
{
  __shared__ __align__(16) bf16 sa[2][128 * 40];
  __shared__ __align__(16) bf16 sb[2][128 * 40];
  const int tid  = threadIdx.x;
  const int lane = tid & 31;
  const int wave = tid >> 5;
  const int wr = wave >> 2;   // 0..1 : 64-row group
  const int wc = wave & 3;    // 0..3 : 32-col group
  const int mb = blockIdx.x * 128;
  const int nb = blockIdx.y * 128;

  // per-thread staging coordinates: 4 slots, rows r0+32i, cols c0..c0+3
  const int r0 = tid >> 3;
  const int c0 = (tid & 7) << 2;

  v8f acc[4][2] = {};
  float4 ra[4], rb[4];

  // prologue: load K-tile 0 and stage into buffer 0
#pragma unroll
  for (int i = 0; i < 4; ++i) {
    ra[i] = *(const float4*)(A + (size_t)(mb + r0 + i * 32) * K + c0);
    rb[i] = *(const float4*)(W + (size_t)(nb + r0 + i * 32) * K + c0);
  }
#pragma unroll
  for (int i = 0; i < 4; ++i) {
    bf16* da = sa[0] + (r0 + i * 32) * 40 + c0;
    da[0] = (bf16)ra[i].x; da[1] = (bf16)ra[i].y; da[2] = (bf16)ra[i].z; da[3] = (bf16)ra[i].w;
    bf16* db = sb[0] + (r0 + i * 32) * 40 + c0;
    db[0] = (bf16)rb[i].x; db[1] = (bf16)rb[i].y; db[2] = (bf16)rb[i].z; db[3] = (bf16)rb[i].w;
  }
  __syncthreads();

  const int nk = K >> 5;
  for (int ik = 0; ik < nk; ++ik) {
    const int cur = ik & 1;

    // issue global loads for the next K-tile (overlap with WMMA below)
    if (ik + 1 < nk) {
      const int kb = (ik + 1) << 5;
#pragma unroll
      for (int i = 0; i < 4; ++i) {
        ra[i] = *(const float4*)(A + (size_t)(mb + r0 + i * 32) * K + kb + c0);
        rb[i] = *(const float4*)(W + (size_t)(nb + r0 + i * 32) * K + kb + c0);
      }
      // L2 prefetch two tiles ahead on the weight stream
      if (ik + 2 < nk)
        __builtin_prefetch(W + (size_t)(nb + r0) * K + ((ik + 2) << 5) + c0, 0, 1);
    }

    // WMMAs on current buffer
    {
      v16bf bfrag[2];
#pragma unroll
      for (int j = 0; j < 2; ++j) bfrag[j] = load_frag(sb[cur], 40, wc * 32 + j * 16, 0, lane);
#pragma unroll
      for (int i = 0; i < 4; ++i) {
        v16bf afrag = load_frag(sa[cur], 40, wr * 64 + i * 16, 0, lane);
#pragma unroll
        for (int j = 0; j < 2; ++j)
          acc[i][j] = __builtin_amdgcn_wmma_f32_16x16x32_bf16(
              false, afrag, false, bfrag[j], (short)0, acc[i][j], false, false);
      }
    }

    // stage next tile into the other buffer
    if (ik + 1 < nk) {
      const int nxt = cur ^ 1;
#pragma unroll
      for (int i = 0; i < 4; ++i) {
        bf16* da = sa[nxt] + (r0 + i * 32) * 40 + c0;
        da[0] = (bf16)ra[i].x; da[1] = (bf16)ra[i].y; da[2] = (bf16)ra[i].z; da[3] = (bf16)ra[i].w;
        bf16* db = sb[nxt] + (r0 + i * 32) * 40 + c0;
        db[0] = (bf16)rb[i].x; db[1] = (bf16)rb[i].y; db[2] = (bf16)rb[i].z; db[3] = (bf16)rb[i].w;
      }
    }
    __syncthreads();
  }

  // epilogue: uniform branches hoisted out of the store loops
  const int g = lane >> 4;
  const int n = lane & 15;
  if (resid) {
#pragma unroll
    for (int i = 0; i < 4; ++i) {
#pragma unroll
      for (int j = 0; j < 2; ++j) {
        const int col  = nb + wc * 32 + j * 16 + n;
        const int row0 = mb + wr * 64 + i * 16 + g * 8;
        const float bv = bias[col];
#pragma unroll
        for (int r = 0; r < 8; ++r) {
          float v = acc[i][j][r] + bv + resid[(size_t)(row0 + r) * N + col];
          C[(size_t)(row0 + r) * N + col] = v;
        }
      }
    }
  } else if (relu) {
#pragma unroll
    for (int i = 0; i < 4; ++i) {
#pragma unroll
      for (int j = 0; j < 2; ++j) {
        const int col  = nb + wc * 32 + j * 16 + n;
        const int row0 = mb + wr * 64 + i * 16 + g * 8;
        const float bv = bias[col];
#pragma unroll
        for (int r = 0; r < 8; ++r)
          C[(size_t)(row0 + r) * N + col] = fmaxf(acc[i][j][r] + bv, 0.f);
      }
    }
  } else {
#pragma unroll
    for (int i = 0; i < 4; ++i) {
#pragma unroll
      for (int j = 0; j < 2; ++j) {
        const int col  = nb + wc * 32 + j * 16 + n;
        const int row0 = mb + wr * 64 + i * 16 + g * 8;
        const float bv = bias[col];
#pragma unroll
        for (int r = 0; r < 8; ++r)
          C[(size_t)(row0 + r) * N + col] = acc[i][j][r] + bv;
      }
    }
  }
}

// ---------------------------------------------------------------------------
// Flash attention: per (head, 64-query tile). 128 threads = 4 waves,
// wave w owns query rows [w*16, w*16+16). Keys streamed in 64-wide tiles.
// S=Q*K^T via WMMA -> LDS f32 -> online softmax -> P(bf16) -> O += P*V via WMMA.
// ---------------------------------------------------------------------------
__global__ __launch_bounds__(128) void flash_kernel(const float* __restrict__ qkv,
                                                    float* __restrict__ att)
{
  __shared__ __align__(16) bf16 sq [64 * 72];
  __shared__ __align__(16) bf16 sk [64 * 72];
  __shared__ __align__(16) bf16 svt[64 * 72];   // V transposed: [feat][key]
  __shared__ __align__(16) bf16 sp_[64 * 72];   // probabilities
  __shared__ float ss[64 * 68];                 // raw scores f32
  __shared__ float m_s[64], l_s[64], osc[64], red[128];

  const int tid  = threadIdx.x;
  const int lane = tid & 31;
  const int wave = tid >> 5;          // 0..3
  const int h  = blockIdx.y;
  const int q0 = blockIdx.x * 64;
  const int g  = lane >> 4;
  const int n  = lane & 15;

  // stage Q (pre-scaled by 1/sqrt(64))
#pragma unroll
  for (int i = 0; i < 8; ++i) {
    int e = tid + i * 128;
    int r = e >> 4;
    int c = (e & 15) << 2;
    float4 v = *(const float4*)(qkv + (size_t)(q0 + r) * 3072 + h * 64 + c);
    bf16* d = sq + r * 72 + c;
    d[0] = (bf16)(v.x * 0.125f); d[1] = (bf16)(v.y * 0.125f);
    d[2] = (bf16)(v.z * 0.125f); d[3] = (bf16)(v.w * 0.125f);
  }
  if (tid < 64) { m_s[tid] = -1e30f; l_s[tid] = 0.f; }
  v8f oacc[4] = {};
  __syncthreads();

  for (int k0 = 0; k0 < SEQ; k0 += 64) {
    // stage K tile and transposed V tile (mask rows >= SEQ)
#pragma unroll
    for (int i = 0; i < 8; ++i) {
      int e = tid + i * 128;
      int r = e >> 4;
      int c = (e & 15) << 2;
      int kr = k0 + r;
      float4 vk = make_float4(0.f, 0.f, 0.f, 0.f);
      float4 vv = make_float4(0.f, 0.f, 0.f, 0.f);
      if (kr < SEQ) {
        vk = *(const float4*)(qkv + (size_t)kr * 3072 + 1024 + h * 64 + c);
        vv = *(const float4*)(qkv + (size_t)kr * 3072 + 2048 + h * 64 + c);
      }
      bf16* dk = sk + r * 72 + c;
      dk[0] = (bf16)vk.x; dk[1] = (bf16)vk.y; dk[2] = (bf16)vk.z; dk[3] = (bf16)vk.w;
      svt[(c + 0) * 72 + r] = (bf16)vv.x;
      svt[(c + 1) * 72 + r] = (bf16)vv.y;
      svt[(c + 2) * 72 + r] = (bf16)vv.z;
      svt[(c + 3) * 72 + r] = (bf16)vv.w;
    }
    __syncthreads();

    // S = Q K^T : wave handles 16 query rows x 64 keys (4 frags, K=64 -> 2 steps)
#pragma unroll
    for (int j = 0; j < 4; ++j) {
      v8f s = {};
#pragma unroll
      for (int kk = 0; kk < 2; ++kk) {
        v16bf a = load_frag(sq, 72, wave * 16, kk * 32, lane);
        v16bf b = load_frag(sk, 72, j * 16,   kk * 32, lane);
        s = __builtin_amdgcn_wmma_f32_16x16x32_bf16(false, a, false, b, (short)0, s, false, false);
      }
      int rbase = wave * 16 + g * 8;
#pragma unroll
      for (int r = 0; r < 8; ++r) ss[(rbase + r) * 68 + j * 16 + n] = s[r];
    }
    __syncthreads();

    // online softmax: 2 threads per row, 32 cols each
    const int row   = tid >> 1;
    const int cbase = (tid & 1) << 5;
    float mold = m_s[row];
    float mx = -1e30f;
#pragma unroll
    for (int c = 0; c < 32; ++c) {
      float xv = ((k0 + cbase + c) < SEQ) ? ss[row * 68 + cbase + c] : -1e30f;
      mx = fmaxf(mx, xv);
    }
    red[tid] = mx;
    __syncthreads();
    float mnew = fmaxf(mold, fmaxf(red[row * 2], red[row * 2 + 1]));
    float sum = 0.f;
#pragma unroll
    for (int c = 0; c < 32; ++c) {
      float pv = 0.f;
      if ((k0 + cbase + c) < SEQ) pv = __expf(ss[row * 68 + cbase + c] - mnew);
      sp_[row * 72 + cbase + c] = (bf16)pv;
      sum += pv;
    }
    __syncthreads();
    red[tid] = sum;
    __syncthreads();
    if ((tid & 1) == 0) {
      float sc = __expf(mold - mnew);
      m_s[row] = mnew;
      l_s[row] = l_s[row] * sc + red[row * 2] + red[row * 2 + 1];
      osc[row] = sc;
    }
    __syncthreads();

    // O = O*scale + P @ V  (4 feat frags, keys 64 -> 2 K-steps)
    float s8[8];
#pragma unroll
    for (int r = 0; r < 8; ++r) s8[r] = osc[wave * 16 + g * 8 + r];
#pragma unroll
    for (int f = 0; f < 4; ++f) {
#pragma unroll
      for (int r = 0; r < 8; ++r) oacc[f][r] *= s8[r];
#pragma unroll
      for (int ks = 0; ks < 2; ++ks) {
        v16bf a = load_frag(sp_, 72, wave * 16, ks * 32, lane);
        v16bf b = load_frag(svt, 72, f * 16,   ks * 32, lane);
        oacc[f] = __builtin_amdgcn_wmma_f32_16x16x32_bf16(false, a, false, b, (short)0, oacc[f], false, false);
      }
    }
    __syncthreads();
  }

  float linv[8];
#pragma unroll
  for (int r = 0; r < 8; ++r) linv[r] = 1.f / l_s[wave * 16 + g * 8 + r];
#pragma unroll
  for (int f = 0; f < 4; ++f)
#pragma unroll
    for (int r = 0; r < 8; ++r)
      att[(size_t)(q0 + wave * 16 + g * 8 + r) * HDIM + h * 64 + f * 16 + n] =
          oacc[f][r] * linv[r];
}

// ---------------------------------------------------------------------------
// Embedding: row 0 = z@Wz^T+bz ; rows 1..2048 = q@Wq^T+bq ; 2049..4096 = p@Wp^T+bp
// + pos + noise_emb[int(mask*100)].  Rows >= SEQ zeroed (GEMM padding).
// ---------------------------------------------------------------------------
__global__ __launch_bounds__(256) void embed_kernel(
    const float* __restrict__ q,  const float* __restrict__ p,
    const float* __restrict__ qm, const float* __restrict__ pm,
    const float* __restrict__ z,
    const float* __restrict__ Wq, const float* __restrict__ bq,
    const float* __restrict__ Wp, const float* __restrict__ bp,
    const float* __restrict__ Wz, const float* __restrict__ bz,
    const float* __restrict__ pos, const float* __restrict__ ne,
    float* __restrict__ x)
{
  const int s = blockIdx.x;
  const int tid = threadIdx.x;
  if (s >= SEQ) {
    for (int j = tid; j < HDIM; j += 256) x[(size_t)s * HDIM + j] = 0.f;
    return;
  }
  const float *src, *W, *b;
  int D; float mask;
  if (s == 0)         { src = z;                           W = Wz; b = bz; D = 128; mask = 1.f; }
  else if (s <= 2048) { src = q + (size_t)(s - 1) * 64;    W = Wq; b = bq; D = 64;  mask = qm[s - 1]; }
  else                { src = p + (size_t)(s - 2049) * 64; W = Wp; b = bp; D = 64;  mask = pm[s - 2049]; }
  const int idx = (int)(mask * 100.f);
  __shared__ float sv[128];
  if (tid < D) sv[tid] = src[tid];
  __syncthreads();
  for (int j = tid; j < HDIM; j += 256) {
    const float* wr = W + (size_t)j * D;
    float acc = b[j];
    for (int d = 0; d < D; ++d) acc += wr[d] * sv[d];
    x[(size_t)s * HDIM + j] = acc + pos[(size_t)s * HDIM + j] + ne[(size_t)idx * HDIM + j];
  }
}

// ---------------------------------------------------------------------------
// LayerNorm over H=1024 (biased variance, eps=1e-5). One block per row.
// ---------------------------------------------------------------------------
__global__ __launch_bounds__(256) void ln_kernel(const float* __restrict__ in,
    const float* __restrict__ w, const float* __restrict__ b, float* __restrict__ out)
{
  __shared__ float red[256];
  const int s = blockIdx.x;
  const int tid = threadIdx.x;
  const float* xr = in + (size_t)s * HDIM;
  float acc = 0.f;
  for (int j = tid; j < HDIM; j += 256) acc += xr[j];
  red[tid] = acc; __syncthreads();
  for (int o = 128; o > 0; o >>= 1) { if (tid < o) red[tid] += red[tid + o]; __syncthreads(); }
  float mean = red[0] * (1.f / HDIM);
  __syncthreads();
  float vs = 0.f;
  for (int j = tid; j < HDIM; j += 256) { float d = xr[j] - mean; vs += d * d; }
  red[tid] = vs; __syncthreads();
  for (int o = 128; o > 0; o >>= 1) { if (tid < o) red[tid] += red[tid + o]; __syncthreads(); }
  float rstd = rsqrtf(red[0] * (1.f / HDIM) + 1e-5f);
  for (int j = tid; j < HDIM; j += 256)
    out[(size_t)s * HDIM + j] = (xr[j] - mean) * rstd * w[j] + b[j];
}

// ---------------------------------------------------------------------------
// Final projection of row 0: out[128] = x[0,:] @ Wout^T + bout
// ---------------------------------------------------------------------------
__global__ __launch_bounds__(128) void final_kernel(const float* __restrict__ x,
    const float* __restrict__ Wout, const float* __restrict__ bout,
    float* __restrict__ out)
{
  const int nn = threadIdx.x;
  const float* wr = Wout + (size_t)nn * HDIM;
  float acc = bout[nn];
  for (int d = 0; d < HDIM; ++d) acc += wr[d] * x[d];
  out[nn] = acc;
}

// ---------------------------------------------------------------------------
extern "C" void kernel_launch(void* const* d_in, const int* in_sizes, int n_in,
                              void* d_out, int out_size, void* d_ws, size_t ws_size,
                              hipStream_t stream) {
  (void)in_sizes; (void)n_in; (void)out_size; (void)ws_size;
  const float* q    = (const float*)d_in[0];
  const float* p    = (const float*)d_in[1];
  const float* qm   = (const float*)d_in[2];
  const float* pm   = (const float*)d_in[3];
  const float* z    = (const float*)d_in[4];
  const float* Wq   = (const float*)d_in[5];
  const float* bq   = (const float*)d_in[6];
  const float* Wp   = (const float*)d_in[7];
  const float* bp   = (const float*)d_in[8];
  const float* Wz   = (const float*)d_in[9];
  const float* bz   = (const float*)d_in[10];
  const float* pos  = (const float*)d_in[11];
  const float* ne   = (const float*)d_in[12];
  const float* Wqkv = (const float*)d_in[13];
  const float* bqkv = (const float*)d_in[14];
  const float* Wo   = (const float*)d_in[15];
  const float* bo   = (const float*)d_in[16];
  const float* ln1w = (const float*)d_in[17];
  const float* ln1b = (const float*)d_in[18];
  const float* W1   = (const float*)d_in[19];
  const float* b1   = (const float*)d_in[20];
  const float* W2   = (const float*)d_in[21];
  const float* b2   = (const float*)d_in[22];
  const float* ln2w = (const float*)d_in[23];
  const float* ln2b = (const float*)d_in[24];
  const float* Wout = (const float*)d_in[25];
  const float* bout = (const float*)d_in[26];

  float* ws  = (float*)d_ws;
  float* x   = ws;                                 // [SP, H]
  float* hb  = ws + (size_t)SP * HDIM;             // [SP, H]
  float* qkv = ws + (size_t)2 * SP * HDIM;         // [SP, 3H]
  float* att = qkv + (size_t)SP * 3 * HDIM;        // [SP, H]
  float* ff  = att + (size_t)SP * HDIM;            // [SP, 4H]

  embed_kernel<<<SP, 256, 0, stream>>>(q, p, qm, pm, z, Wq, bq, Wp, bp, Wz, bz, pos, ne, x);

  for (int i = 0; i < 6; ++i) {
    gemm_kernel<<<dim3(SP / 128, 24), 256, 0, stream>>>(
        x, Wqkv + (size_t)i * 3072 * 1024, bqkv + (size_t)i * 3072, nullptr,
        qkv, SP, 3072, 1024, 0);
    flash_kernel<<<dim3(SP / 64, NHEADS), 128, 0, stream>>>(qkv, att);
    gemm_kernel<<<dim3(SP / 128, 8), 256, 0, stream>>>(
        att, Wo + (size_t)i * 1024 * 1024, bo + (size_t)i * 1024, x,
        hb, SP, 1024, 1024, 0);
    ln_kernel<<<SEQ, 256, 0, stream>>>(hb, ln1w + (size_t)i * 1024, ln1b + (size_t)i * 1024, x);
    gemm_kernel<<<dim3(SP / 128, 32), 256, 0, stream>>>(
        x, W1 + (size_t)i * 4096 * 1024, b1 + (size_t)i * 4096, nullptr,
        ff, SP, 4096, 1024, 1);
    gemm_kernel<<<dim3(SP / 128, 8), 256, 0, stream>>>(
        ff, W2 + (size_t)i * 1024 * 4096, b2 + (size_t)i * 1024, x,
        hb, SP, 1024, 4096, 0);
    ln_kernel<<<SEQ, 256, 0, stream>>>(hb, ln2w + (size_t)i * 1024, ln2b + (size_t)i * 1024, x);
  }

  final_kernel<<<1, 128, 0, stream>>>(x, Wout, bout, (float*)d_out);
}
